// GateRetention_77498389889493
// MI455X (gfx1250) — matrix operand order repaired and checked by reading, not compiled, not run on hardware
//
#include <hip/hip_runtime.h>

// ---------------------------------------------------------------------------
// Types for CDNA5 WMMA (wave32)
// ---------------------------------------------------------------------------
typedef __bf16 bf16_t;
typedef bf16_t v16bf __attribute__((ext_vector_type(16)));
typedef float  v8f   __attribute__((ext_vector_type(8)));
typedef unsigned int v8u __attribute__((ext_vector_type(8)));
typedef unsigned int u32x4 __attribute__((ext_vector_type(4)));
typedef int i32x4 __attribute__((ext_vector_type(4)));
typedef int i32x8 __attribute__((ext_vector_type(8)));

#define DEV __device__ __forceinline__

#if defined(__has_builtin)
#if __has_builtin(__builtin_amdgcn_tensor_load_to_lds)
#define HAVE_TDM 1
#endif
#endif
#ifndef HAVE_TDM
#define HAVE_TDM 0
#endif

DEV unsigned short f2bf(float f) {
  unsigned int u = __builtin_bit_cast(unsigned int, f);
  u += 0x7FFFu + ((u >> 16) & 1u);          // round-to-nearest-even
  return (unsigned short)(u >> 16);
}
DEV float bf2f(unsigned short h) {
  return __builtin_bit_cast(float, ((unsigned int)h) << 16);
}
DEV unsigned int pack2(float lo, float hi) {
  return (unsigned int)f2bf(lo) | ((unsigned int)f2bf(hi) << 16);
}
DEV v8f vzero8() {
  v8f z;
#pragma unroll
  for (int i = 0; i < 8; ++i) z[i] = 0.f;
  return z;
}
DEV v8f wmma_bf16(v8u a, v8u b, v8f c) {
  return __builtin_amdgcn_wmma_f32_16x16x32_bf16(
      false, __builtin_bit_cast(v16bf, a),
      false, __builtin_bit_cast(v16bf, b),
      (short)0, c, false, false);
}

// ---- CDNA5 async global->LDS copy (ASYNCcnt) ------------------------------
DEV void async_load_b128(unsigned lds_off, const void* gptr) {
  asm volatile("global_load_async_to_lds_b128 %0, %1, off"
               :: "v"(lds_off), "v"((unsigned long long)(size_t)gptr)
               : "memory");
}
DEV void wait_asynccnt0() { asm volatile("s_wait_asynccnt 0" ::: "memory"); }

// ---- CDNA5 Tensor Data Mover: 256x128 bf16 tile, row stride 2048 elems,
//      LDS row pad 16B (-> 136-ushort rows).  Issued once per wave.
#if HAVE_TDM
DEV void tdm_load_tile_256x128(const unsigned short* dst_lds,
                               const unsigned short* gsrc) {
  unsigned lds = (unsigned)(size_t)dst_lds;
  unsigned long long ga = (unsigned long long)(size_t)gsrc;
  u32x4 g0;
  g0[0] = 1u;                                            // count=1, no gather
  g0[1] = lds;                                           // lds_addr (bytes)
  g0[2] = (unsigned)(ga & 0xFFFFFFFFu);                  // global_addr[31:0]
  g0[3] = (unsigned)((ga >> 32) & 0x01FFFFFFu) | 0x80000000u; // [56:32] | type=2
  i32x8 g1;
  g1[0] = (int)((1u << 16)      // data_size = 2B
              | (1u << 20)      // pad_enable
              | (5u << 22)      // pad_interval: 64 DWORDs (=256B row)
              | (3u << 25));    // pad_amount:   4 DWORDs (=16B)
  g1[1] = (int)(128u << 16);    // tensor_dim0 = 128 (elements)
  g1[2] = (int)(256u << 16);    // tensor_dim1 = 256 (rows)
  g1[3] = (int)(128u << 16);    // tile_dim0 = 128
  g1[4] = 256;                  // tile_dim1 = 256, tile_dim2 = 0
  g1[5] = 2048;                 // tensor_dim0_stride = 2048 elements
  g1[6] = 0; g1[7] = 0;
  i32x4 z4 = {0, 0, 0, 0};
#if __clang_major__ >= 23
  i32x8 z8 = {0, 0, 0, 0, 0, 0, 0, 0};
  __builtin_amdgcn_tensor_load_to_lds(g0, g1, z4, z4, z8, 0);
#else
  __builtin_amdgcn_tensor_load_to_lds(g0, g1, z4, z4, 0);
#endif
}
#endif

// ---------------------------------------------------------------------------
// Fragment loaders (wave32 layouts per CDNA5 ISA 7.12.2)
// A 16x32 bf16: lanes 0-15 -> M=lane, K in {0..7,16..23}; lanes 16-31 -> K in {8..15,24..31}
// B 32x16 bf16 (from [N][K]-contiguous LDS): lanes 0-15 -> N=lane, K in {0..7,16..23}; lanes 16-31 -> K in {8..15,24..31}
// C/D 16x16 f32: VGPR r -> M=r (lanes 0-15) / M=r+8 (lanes 16-31), N=lane%16
// ---------------------------------------------------------------------------
DEV v8u load_afrag(const unsigned short* p, int ld, int row0, int col0, int lane) {
  int hlf = lane >> 4, m = lane & 15;
  const unsigned short* r = p + (row0 + m) * ld + col0 + (hlf ? 8 : 0);
  uint4 a = *(const uint4*)r;
  uint4 b = *(const uint4*)(r + 16);
  v8u f;
  f[0]=a.x; f[1]=a.y; f[2]=a.z; f[3]=a.w;
  f[4]=b.x; f[5]=b.y; f[6]=b.z; f[7]=b.w;
  return f;
}
DEV v8u load_afrag_f32(const float* p, int ld, int row0, int col0, int lane) {
  int hlf = lane >> 4, m = lane & 15;
  const float* r = p + (row0 + m) * ld + col0 + (hlf ? 8 : 0);
  v8u f;
#pragma unroll
  for (int i = 0; i < 4; ++i) f[i] = pack2(r[2*i], r[2*i+1]);
#pragma unroll
  for (int i = 0; i < 4; ++i) f[4+i] = pack2(r[16+2*i], r[16+2*i+1]);
  return f;
}
DEV v8u load_afrag_scaled(const unsigned short* p, int ld, int row0, int col0,
                          float s, int lane) {
  int hlf = lane >> 4, m = lane & 15;
  const unsigned short* r = p + (row0 + m) * ld + col0 + (hlf ? 8 : 0);
  v8u f;
#pragma unroll
  for (int i = 0; i < 4; ++i)
    f[i] = pack2(bf2f(r[2*i]) * s, bf2f(r[2*i+1]) * s);
#pragma unroll
  for (int i = 0; i < 4; ++i)
    f[4+i] = pack2(bf2f(r[16+2*i]) * s, bf2f(r[16+2*i+1]) * s);
  return f;
}
DEV v8u load_bfrag(const unsigned short* p, int ld, int n0, int k0, int lane) {
  int hlf = lane >> 4, n = lane & 15;
  const unsigned short* r = p + (n0 + n) * ld + k0 + hlf * 8;
  uint4 a = *(const uint4*)r;
  uint4 b = *(const uint4*)(r + 16);
  v8u f;
  f[0]=a.x; f[1]=a.y; f[2]=a.z; f[3]=a.w;
  f[4]=b.x; f[5]=b.y; f[6]=b.z; f[7]=b.w;
  return f;
}
DEV v8u load_bfrag_f32(const float* p, int ld, int n0, int k0, int lane) {
  int hlf = lane >> 4, n = lane & 15;
  const float* r = p + (n0 + n) * ld + k0 + hlf * 8;
  v8u f;
#pragma unroll
  for (int i = 0; i < 8; ++i) f[i] = pack2(r[2*i + (i >= 4 ? 8 : 0)], r[2*i + 1 + (i >= 4 ? 8 : 0)]);
  return f;
}
// kd^T A-fragment: M=d gathered from sK[m][d] with per-m decay scale etb[m]
DEV v8u load_kdT_frag(const unsigned short* sK, int ld, int d0, int m0,
                      const float* etb, int lane) {
  int hlf = lane >> 4, dl = lane & 15;
  int col = d0 + dl;
  v8u f;
#pragma unroll
  for (int i = 0; i < 4; ++i) {
    int mA = m0 + (hlf ? 8 : 0) + 2*i;
    f[i] = pack2(bf2f(sK[mA*ld + col]) * etb[mA],
                 bf2f(sK[(mA+1)*ld + col]) * etb[mA+1]);
  }
#pragma unroll
  for (int i = 0; i < 4; ++i) {
    int mA = m0 + 16 + (hlf ? 8 : 0) + 2*i;
    f[4+i] = pack2(bf2f(sK[mA*ld + col]) * etb[mA],
                   bf2f(sK[(mA+1)*ld + col]) * etb[mA+1]);
  }
  return f;
}

// ---------------------------------------------------------------------------
// Kernel 1: bf16-WMMA GEMM, f32 in/out.  A tile staged via async global->LDS
// (f32, convert at fragment build); W tile converted + transposed manually.
// C[M,N] = A[M,K] * W[K,N].  BM=BN=128, BK=32, 256 threads (8 waves 4x2).
// ---------------------------------------------------------------------------
__global__ __launch_bounds__(256) void wmma_gemm_kernel(
    const float* __restrict__ A, const float* __restrict__ W,
    float* __restrict__ C, int M, int N, int K) {
  __shared__ float sAf[128 * 40];            // [m][k] f32, pad 8
  __shared__ unsigned short sBT[128 * 40];   // [n][k] bf16, pad 8
  const int tid = threadIdx.x, lane = tid & 31, wave = tid >> 5;
  const int wm = wave >> 1, wn = wave & 1;
  const int hlf = lane >> 4, nl = lane & 15;
  const long m0 = (long)blockIdx.y * 128, n0 = (long)blockIdx.x * 128;
  v8f acc[2][4];
#pragma unroll
  for (int i = 0; i < 2; ++i)
#pragma unroll
    for (int j = 0; j < 4; ++j) acc[i][j] = vzero8();

  const int am = tid >> 1, ak = (tid & 1) * 16;       // A: 128 rows x 32k
  const int bk = tid >> 3, bn = (tid & 7) * 16;       // W: 32 rows x 128n
  const unsigned ldsA = (unsigned)(size_t)(sAf + am * 40 + ak);

  for (int k0 = 0; k0 < K; k0 += 32) {
    {  // A tile: 16 floats/thread via 4 async b128 copies
      const float* gp = A + (m0 + am) * K + k0 + ak;
      async_load_b128(ldsA +  0, gp +  0);
      async_load_b128(ldsA + 16, gp +  4);
      async_load_b128(ldsA + 32, gp +  8);
      async_load_b128(ldsA + 48, gp + 12);
    }
    {  // W tile: convert f32->bf16, store transposed [n][k]
      const float4* bp = (const float4*)(W + (long)(k0 + bk) * N + n0 + bn);
      if (k0 + 64 < K) __builtin_prefetch(W + (long)(k0 + 64 + bk) * N + n0 + bn, 0, 1);
      float t1[16];
      *(float4*)(t1+0)=bp[0]; *(float4*)(t1+4)=bp[1];
      *(float4*)(t1+8)=bp[2]; *(float4*)(t1+12)=bp[3];
#pragma unroll
      for (int j = 0; j < 16; ++j) sBT[(bn + j)*40 + bk] = f2bf(t1[j]);
    }
    wait_asynccnt0();
    __syncthreads();
    v8u af0 = load_afrag_f32(sAf, 40, wm*32,      0, lane);
    v8u af1 = load_afrag_f32(sAf, 40, wm*32 + 16, 0, lane);
#pragma unroll
    for (int tj = 0; tj < 4; ++tj) {
      v8u bf_ = load_bfrag(sBT, 40, wn*64 + tj*16, 0, lane);
      acc[0][tj] = wmma_bf16(af0, bf_, acc[0][tj]);
      acc[1][tj] = wmma_bf16(af1, bf_, acc[1][tj]);
    }
    __syncthreads();
  }
#pragma unroll
  for (int ti = 0; ti < 2; ++ti)
#pragma unroll
    for (int tj = 0; tj < 4; ++tj)
#pragma unroll
      for (int r = 0; r < 8; ++r) {
        long row = m0 + wm*32 + ti*16 + r + 8*hlf;
        long col = n0 + wn*64 + tj*16 + nl;
        C[row * N + col] = acc[ti][tj][r];
      }
}

// ---------------------------------------------------------------------------
// Kernel 2: gt = log_sigmoid((x+c) @ Wgt) / 16    -> [B*T, 16]
// ---------------------------------------------------------------------------
__global__ __launch_bounds__(256) void gt_project_kernel(
    const float* __restrict__ x, const float* __restrict__ c,
    const float* __restrict__ Wgt, float* __restrict__ gt) {
  const int wave = threadIdx.x >> 5, lane = threadIdx.x & 31;
  const long row = (long)blockIdx.x * 8 + wave;
  const float* xr = x + row * 2048;
  const float* cr = c + row * 2048;
  float acc[16];
#pragma unroll
  for (int i = 0; i < 16; ++i) acc[i] = 0.f;
  for (int k = lane; k < 2048; k += 32) {
    float xv = xr[k] + cr[k];
    const float4* wr = (const float4*)(Wgt + k * 16);
    float4 w0 = wr[0], w1 = wr[1], w2 = wr[2], w3 = wr[3];
    acc[0]+=xv*w0.x; acc[1]+=xv*w0.y; acc[2]+=xv*w0.z; acc[3]+=xv*w0.w;
    acc[4]+=xv*w1.x; acc[5]+=xv*w1.y; acc[6]+=xv*w1.z; acc[7]+=xv*w1.w;
    acc[8]+=xv*w2.x; acc[9]+=xv*w2.y; acc[10]+=xv*w2.z; acc[11]+=xv*w2.w;
    acc[12]+=xv*w3.x; acc[13]+=xv*w3.y; acc[14]+=xv*w3.z; acc[15]+=xv*w3.w;
  }
#pragma unroll
  for (int off = 1; off < 32; off <<= 1)
#pragma unroll
    for (int i = 0; i < 16; ++i) acc[i] += __shfl_xor(acc[i], off, 32);
  if (lane == 0) {
#pragma unroll
    for (int i = 0; i < 16; ++i) {
      float z = acc[i];
      float ls = fminf(z, 0.f) - log1pf(__expf(-fabsf(z)));
      gt[row * 16 + i] = ls * (1.f / 16.f);
    }
  }
}

// ---------------------------------------------------------------------------
// Kernel 3: per-(b,h,chunk) inclusive cumsum of gt -> bcum, chunk total
// ---------------------------------------------------------------------------
__global__ __launch_bounds__(256) void cumsum_kernel(
    const float* __restrict__ gt, float* __restrict__ bcum,
    float* __restrict__ total) {
  __shared__ float sbuf[256];
  const int bh = blockIdx.x >> 4, nc = blockIdx.x & 15;
  const int b = bh >> 4, h = bh & 15;
  const int t0 = nc * 256, tid = threadIdx.x;
  sbuf[tid] = gt[((long)(b * 4096 + t0 + tid)) * 16 + h];
  __syncthreads();
  for (int off = 1; off < 256; off <<= 1) {
    float t = (tid >= off) ? sbuf[tid - off] : 0.f;
    __syncthreads();
    sbuf[tid] += t;
    __syncthreads();
  }
  bcum[(long)bh * 4096 + t0 + tid] = sbuf[tid];
  if (tid == 255) total[bh * 16 + nc] = sbuf[255];
}

// ---------------------------------------------------------------------------
// Kernel 4: RMSNorm over dim 2048 (f32 in -> bf16 out, optional extra scale)
// ---------------------------------------------------------------------------
__global__ __launch_bounds__(256) void rmsnorm_bf16_kernel(
    const float* __restrict__ in, unsigned short* __restrict__ out, float scale) {
  __shared__ float red[256];
  const long row = blockIdx.x;
  const float* r = in + row * 2048;
  const int tid = threadIdx.x;
  float s = 0.f;
  for (int i = tid; i < 2048; i += 256) { float xv = r[i]; s += xv * xv; }
  red[tid] = s; __syncthreads();
  for (int off = 128; off > 0; off >>= 1) {
    if (tid < off) red[tid] += red[tid + off];
    __syncthreads();
  }
  float rn = rsqrtf(red[0] * (1.f / 2048.f) + 1e-5f) * scale;
  for (int i = tid; i < 2048; i += 256) out[row * 2048 + i] = f2bf(r[i] * rn);
}

// ---------------------------------------------------------------------------
// Kernel 5: chunkwise gated retention.  One block (16 waves) per (b,h).
// q/k chunk tiles staged by the Tensor Data Mover (padded rows), v staged
// transposed manually.  State S in LDS (f32, e-major).  All matmuls on WMMA.
// ---------------------------------------------------------------------------
__global__ __launch_bounds__(512) void retention_kernel(
    const unsigned short* __restrict__ qn, const unsigned short* __restrict__ kn,
    const unsigned short* __restrict__ vn, const float* __restrict__ bcum,
    const float* __restrict__ total, float* __restrict__ o) {
  extern __shared__ char smem[];
  float*          ST  = (float*)smem;                           // [128][136] f32 (ST[e][d])
  unsigned short* sQ  = (unsigned short*)(smem + 69632);        // [256][136] bf16 (q*scale)
  unsigned short* sK  = (unsigned short*)(smem + 139264);       // [256][136] bf16
  unsigned short* sVT = (unsigned short*)(smem + 208896);       // [128][264] bf16 (v^T)
  unsigned short* bounce = (unsigned short*)(smem + 276480);    // 16 x [16][32]
  float* sB   = (float*)(smem + 292864);                        // [256] cum log-decay
  float* sEb  = (float*)(smem + 293888);                        // exp(b)
  float* sEtb = (float*)(smem + 294912);                        // exp(total - b)
  __shared__ float s_exptot;

  const int tid = threadIdx.x, lane = tid & 31, wave = tid >> 5;
  const int hlf = lane >> 4, nl = lane & 15;
  const int bh = blockIdx.x;                 // b*16 + h
  const int b = bh >> 4, h = bh & 15;
  const int rb = wave * 16;                  // this wave's output row band
  unsigned short* wb = bounce + wave * (16 * 32);
  const int dtile = wave >> 1;               // state d-tile owned by this wave

  for (int i = tid; i < 128 * 136; i += 512) ST[i] = 0.f;

  for (int nc = 0; nc < 16; ++nc) {
    const int t0 = nc * 256;
    const long base = ((long)(b * 4096 + t0)) * 2048 + h * 128;
    // ---- stage chunk tiles into LDS ----
#if HAVE_TDM
    if (wave == 0) {
      tdm_load_tile_256x128(sQ, qn + base);   // TDM handles pad -> 136 stride
      tdm_load_tile_256x128(sK, kn + base);
    }
#endif
    for (int i = tid; i < 4096; i += 512) {
      int row = i >> 4, seg = (i & 15) << 3;
#if !HAVE_TDM
      *(uint4*)(sQ + row * 136 + seg) = *(const uint4*)(qn + base + (long)row * 2048 + seg);
      *(uint4*)(sK + row * 136 + seg) = *(const uint4*)(kn + base + (long)row * 2048 + seg);
#endif
      uint4 vv = *(const uint4*)(vn + base + (long)row * 2048 + seg);
      unsigned short tmp[8]; *(uint4*)tmp = vv;
#pragma unroll
      for (int j = 0; j < 8; ++j) sVT[(seg + j) * 264 + row] = tmp[j];
    }
    if (tid < 256) {
      float bv = bcum[(long)bh * 4096 + t0 + tid];
      sB[tid] = bv; sEb[tid] = __expf(bv);
      float tv = total[bh * 16 + nc];
      sEtb[tid] = __expf(tv - bv);
      if (tid == 0) s_exptot = __expf(tv);
    }
#if HAVE_TDM
    if (wave == 0) __builtin_amdgcn_s_wait_tensorcnt(0);
#endif
    __syncthreads();

    // ---- inter-chunk: out = (q*scale*exp(b)) @ S ----
    v8f acc[8];
#pragma unroll
    for (int n = 0; n < 8; ++n) acc[n] = vzero8();
    {
      float rowscale = sEb[rb + nl];
#pragma unroll
      for (int ks = 0; ks < 4; ++ks) {
        v8u af = load_afrag_scaled(sQ, 136, rb, ks * 32, rowscale, lane);
#pragma unroll
        for (int n = 0; n < 8; ++n)
          acc[n] = wmma_bf16(af, load_bfrag_f32(ST, 136, n * 16, ks * 32, lane), acc[n]);
      }
    }

    // ---- intra-chunk: A = q k^T, causal decay mask, out += A v ----
    const int jsMax = rb >> 5;
    for (int js = 0; js <= jsMax; ++js) {
      v8f a2[2]; a2[0] = vzero8(); a2[1] = vzero8();
#pragma unroll
      for (int ks = 0; ks < 4; ++ks) {
        v8u af = load_afrag(sQ, 136, rb, ks * 32, lane);
        a2[0] = wmma_bf16(af, load_bfrag(sK, 136, (2*js    ) * 16, ks * 32, lane), a2[0]);
        a2[1] = wmma_bf16(af, load_bfrag(sK, 136, (2*js + 1) * 16, ks * 32, lane), a2[1]);
      }
#pragma unroll
      for (int t = 0; t < 2; ++t) {
        int m = (2*js + t) * 16 + nl;
#pragma unroll
        for (int r = 0; r < 8; ++r) {
          int cc = rb + r + 8 * hlf;
          float val = (cc >= m) ? a2[t][r] * __expf(sB[cc] - sB[m]) : 0.f;
          wb[(r + 8 * hlf) * 32 + t * 16 + nl] = f2bf(val);
        }
      }
      asm volatile("s_wait_dscnt 0" ::: "memory");   // wave-private LDS RAW
      v8u af = load_afrag(wb, 32, 0, 0, lane);
#pragma unroll
      for (int n = 0; n < 8; ++n)
        acc[n] = wmma_bf16(af, load_bfrag(sVT, 264, n * 16, js * 32, lane), acc[n]);
    }

    // ---- subln (rms over head_dim=128) + store ----
    float inv[8];
#pragma unroll
    for (int r = 0; r < 8; ++r) {
      float s = 0.f;
#pragma unroll
      for (int n = 0; n < 8; ++n) s += acc[n][r] * acc[n][r];
      s += __shfl_xor(s, 1, 32); s += __shfl_xor(s, 2, 32);
      s += __shfl_xor(s, 4, 32); s += __shfl_xor(s, 8, 32);
      inv[r] = rsqrtf(s * (1.f / 128.f) + 1e-5f);
    }
    {
      long ob = ((long)bh * 4096 + t0 + rb) * 128;
#pragma unroll
      for (int n = 0; n < 8; ++n)
#pragma unroll
        for (int r = 0; r < 8; ++r)
          o[ob + (long)(r + 8 * hlf) * 128 + n * 16 + nl] = acc[n][r] * inv[r];
    }

    // ---- state update: S = exp(total)*S + (k*exp(total-b))^T @ v ----
    v8f sacc[4];
#pragma unroll
    for (int i = 0; i < 4; ++i) {
      int e0 = ((wave * 4 + i) & 7) * 16;
#pragma unroll
      for (int r = 0; r < 8; ++r)
        sacc[i][r] = s_exptot * ST[(e0 + nl) * 136 + dtile * 16 + 8 * hlf + r];
    }
#pragma unroll
    for (int ks = 0; ks < 8; ++ks) {
      v8u af = load_kdT_frag(sK, 136, dtile * 16, ks * 32, sEtb, lane);
#pragma unroll
      for (int i = 0; i < 4; ++i) {
        int e0 = ((wave * 4 + i) & 7) * 16;
        sacc[i] = wmma_bf16(af, load_bfrag(sVT, 264, e0, ks * 32, lane), sacc[i]);
      }
    }
    __syncthreads();   // everyone done reading old S
#pragma unroll
    for (int i = 0; i < 4; ++i) {
      int e0 = ((wave * 4 + i) & 7) * 16;
#pragma unroll
      for (int r = 0; r < 8; ++r)
        ST[(e0 + nl) * 136 + dtile * 16 + 8 * hlf + r] = sacc[i][r];
    }
    __syncthreads();   // new S visible; LDS tiles free for next chunk
  }
}

// ---------------------------------------------------------------------------
// Kernel 6: act = silu(g) * o  (o is [B,H,T,128] -> gather to [B,T,2048])
// ---------------------------------------------------------------------------
__global__ __launch_bounds__(256) void swiglu_kernel(
    const float* __restrict__ g, const float* __restrict__ o,
    float* __restrict__ act) {
  const long nTot = (long)2 * 4096 * 2048;
  for (long idx = (long)blockIdx.x * blockDim.x + threadIdx.x; idx < nTot;
       idx += (long)gridDim.x * blockDim.x) {
    int e = (int)(idx & 127);
    int h = (int)((idx >> 7) & 15);
    long bt = idx >> 11;
    long bb = bt >> 12, t = bt & 4095;
    float ov = o[(((bb * 16 + h) << 12) + t) * 128 + e];
    float gv = g[idx];
    act[idx] = gv / (1.f + __expf(-gv)) * ov;
  }
}

// ---------------------------------------------------------------------------
// Host orchestration
// ---------------------------------------------------------------------------
extern "C" void kernel_launch(void* const* d_in, const int* in_sizes, int n_in,
                              void* d_out, int out_size, void* d_ws, size_t ws_size,
                              hipStream_t stream) {
  (void)in_sizes; (void)n_in; (void)out_size; (void)ws_size;
  const float* x   = (const float*)d_in[0];
  const float* c   = (const float*)d_in[1];
  const float* Wq  = (const float*)d_in[2];
  const float* Wk  = (const float*)d_in[3];
  const float* Wv  = (const float*)d_in[4];
  const float* Wg  = (const float*)d_in[5];
  const float* Wgt = (const float*)d_in[6];
  const float* Wo  = (const float*)d_in[7];
  float* out = (float*)d_out;

  char* ws = (char*)d_ws;
  const size_t SZ_F = (size_t)8192 * 2048 * 4;
  const size_t SZ_H = (size_t)8192 * 2048 * 2;
  float* qf = (float*)ws;                 ws += SZ_F;
  float* kf = (float*)ws;                 ws += SZ_F;
  float* vf = (float*)ws;                 ws += SZ_F;
  float* gf = (float*)ws;                 ws += SZ_F;
  float* gtb    = (float*)ws;             ws += (size_t)8192 * 16 * 4;
  float* bcum   = (float*)ws;             ws += (size_t)32 * 4096 * 4;
  float* totalb = (float*)ws;             ws += 4096;
  unsigned short* qh = (unsigned short*)ws; ws += SZ_H;
  unsigned short* kh = (unsigned short*)ws; ws += SZ_H;
  unsigned short* vh = (unsigned short*)ws; ws += SZ_H;
  float* ob   = kf;   // alias: kf dead after rmsnorm(k)
  float* actb = qf;   // alias: qf dead after rmsnorm(q)

  dim3 gg(16, 64);
  wmma_gemm_kernel<<<gg, 256, 0, stream>>>(x, Wq, qf, 8192, 2048, 2048);
  wmma_gemm_kernel<<<gg, 256, 0, stream>>>(x, Wk, kf, 8192, 2048, 2048);
  wmma_gemm_kernel<<<gg, 256, 0, stream>>>(x, Wv, vf, 8192, 2048, 2048);
  wmma_gemm_kernel<<<gg, 256, 0, stream>>>(x, Wg, gf, 8192, 2048, 2048);
  gt_project_kernel<<<1024, 256, 0, stream>>>(x, c, Wgt, gtb);
  cumsum_kernel<<<512, 256, 0, stream>>>(gtb, bcum, totalb);
  rmsnorm_bf16_kernel<<<8192, 256, 0, stream>>>(qf, qh, 0.08838834764831845f); // 1/sqrt(128)
  rmsnorm_bf16_kernel<<<8192, 256, 0, stream>>>(kf, kh, 1.0f);
  rmsnorm_bf16_kernel<<<8192, 256, 0, stream>>>(vf, vh, 1.0f);
  hipFuncSetAttribute(reinterpret_cast<const void*>(retention_kernel),
                      hipFuncAttributeMaxDynamicSharedMemorySize, 295936);
  retention_kernel<<<32, 512, 295936, stream>>>(qh, kh, vh, bcum, totalb, ob);
  swiglu_kernel<<<2048, 256, 0, stream>>>(gf, ob, actb);
  wmma_gemm_kernel<<<gg, 256, 0, stream>>>(actb, Wo, out, 8192, 2048, 2048);
}
